// ST_BLOCK_5_19447611916424
// MI455X (gfx1250) — matrix operand order repaired and checked by reading, not compiled, not run on hardware
//
#include <hip/hip_runtime.h>
#include <stdint.h>

#define Bd   16
#define CIN  32
#define COUT 64
#define Nd   1024
#define Td   64
#define FEAT 96    // CIN+COUT
#define GATE 256   // 4*COUT
#define FK   288   // FEAT*K

typedef __attribute__((ext_vector_type(16))) __bf16 v16bf;
typedef __attribute__((ext_vector_type(8)))  float  v8f;

union TileU { v16bf v; uint4 u[2]; };

__device__ __forceinline__ unsigned short f2bf(float x) {
  union { float f; unsigned u; } p; p.f = x;
  unsigned r = p.u + 0x7FFFu + ((p.u >> 16) & 1u);
  return (unsigned short)(r >> 16);
}

__device__ __forceinline__ uint4 pack8(v8f a) {
  uint4 o;
  o.x = (unsigned)f2bf(a[0]) | ((unsigned)f2bf(a[1]) << 16);
  o.y = (unsigned)f2bf(a[2]) | ((unsigned)f2bf(a[3]) << 16);
  o.z = (unsigned)f2bf(a[4]) | ((unsigned)f2bf(a[5]) << 16);
  o.w = (unsigned)f2bf(a[6]) | ((unsigned)f2bf(a[7]) << 16);
  return o;
}

// A-matrix 16x32 bf16 tile, contiguous k order (same as load_tile_B):
// lanes 0-15: row = row0+lane,  K = [0..15]
// lanes 16-31: row = row0+lane-16, K = [16..31]
__device__ __forceinline__ v16bf load_tile_A(const unsigned short* base, int stride,
                                             int row0, int k0, int lane) {
  int r = row0 + (lane & 15);
  int o = (lane < 16) ? 0 : 16;
  const uint4* p = reinterpret_cast<const uint4*>(base + (size_t)r * stride + k0 + o);
  TileU t;
  t.u[0] = p[0];   // K = o .. o+7
  t.u[1] = p[1];   // K = o+8 .. o+15
  return t.v;
}

// B-matrix 32x16 bf16 tile; memory holds Bmem[col][k] contiguous in k.
// lanes 0-15: col = col0+lane,  K = 0..15 ; lanes 16-31: col-16, K = 16..31
__device__ __forceinline__ v16bf load_tile_B(const unsigned short* base, int stride,
                                             int col0, int k0, int lane) {
  int c  = col0 + (lane & 15);
  int kb = k0 + ((lane < 16) ? 0 : 16);
  const uint4* p = reinterpret_cast<const uint4*>(base + (size_t)c * stride + kb);
  TileU t;
  t.u[0] = p[0];
  t.u[1] = p[1];
  return t.v;
}

__device__ __forceinline__ v8f wmma_bf(v16bf a, v16bf b, v8f c) {
  return __builtin_amdgcn_wmma_f32_16x16x32_bf16(false, a, false, b,
                                                 (short)0, c, false, false);
}

__device__ __forceinline__ float sgm(float x) { return 1.0f / (1.0f + __expf(-x)); }

// ---------------- setup kernels ----------------

__global__ void k_setup_adj(const float* __restrict__ adj,
                            unsigned short* __restrict__ adjbf,
                            unsigned short* __restrict__ adjT) {
  int idx = blockIdx.x * blockDim.x + threadIdx.x;
  if (idx >= Nd * Nd) return;
  int q = idx >> 10, n = idx & 1023;
  unsigned short v = f2bf(adj[idx]);
  adjbf[idx] = v;             // L1, row-major [q][n]
  adjT[(size_t)n * Nd + q] = v;
}

__global__ void k_setup_w(const float* __restrict__ W, unsigned short* __restrict__ Wre) {
  int idx = blockIdx.x * blockDim.x + threadIdx.x;
  if (idx >= GATE * FK) return;
  int o = idx / FK, fk = idx % FK;
  int k = fk / FEAT, f = fk % FEAT;
  Wre[idx] = f2bf(W[(size_t)o * FK + f * 3 + k]);   // W row layout is (f,k) k-minor
}

__global__ void k_init(float* __restrict__ cstate, unsigned short* __restrict__ tem,
                       unsigned short* __restrict__ gcatT) {
  int idx = blockIdx.x * blockDim.x + threadIdx.x;
  if (idx >= Bd * COUT * Nd) return;
  int b  = idx / (COUT * Nd);
  int r  = idx % (COUT * Nd);
  int co = r / Nd, n = r % Nd;
  cstate[idx] = 0.0f;
  tem[((size_t)b * FEAT + CIN + co) * Nd + n] = 0;     // h rows of tem
  gcatT[((size_t)b * Nd + n) * FK + CIN + co] = 0;     // h cols of gcatT
}

// L2 = 2*adj@adj - I  (bf16 WMMA, f32 accum)
__global__ void k_cheb2(const unsigned short* __restrict__ adjbf,
                        const unsigned short* __restrict__ adjT,
                        unsigned short* __restrict__ L2) {
  int lane = threadIdx.x & 31;
  int wave = threadIdx.x >> 5;
  int q0 = blockIdx.x * 128 + wave * 16;
  int m0 = blockIdx.y * 64;
  v8f acc[4] = {};
  for (int k0 = 0; k0 < Nd; k0 += 32) {
    v16bf at = load_tile_A(adjbf, Nd, q0, k0, lane);
    #pragma unroll
    for (int ct = 0; ct < 4; ++ct) {
      v16bf bt = load_tile_B(adjT, Nd, m0 + ct * 16, k0, lane);
      acc[ct] = wmma_bf(at, bt, acc[ct]);
    }
  }
  #pragma unroll
  for (int ct = 0; ct < 4; ++ct)
    #pragma unroll
    for (int r = 0; r < 8; ++r) {
      int q = q0 + r + ((lane < 16) ? 0 : 8);
      int m = m0 + ct * 16 + (lane & 15);
      float v = 2.0f * acc[ct][r] - ((q == m) ? 1.0f : 0.0f);
      L2[(size_t)q * Nd + m] = f2bf(v);
    }
}

// ---------------- per-timestep kernels ----------------

// Scatter x_t (bf16) into tem rows [0,32) and gcatT cols [0,32)
__global__ void k_prep(const float* __restrict__ x, unsigned short* __restrict__ tem,
                       unsigned short* __restrict__ gcatT, int t) {
  int idx = blockIdx.x * blockDim.x + threadIdx.x;
  if (idx >= Bd * CIN * Nd) return;
  int b = idx / (CIN * Nd);
  int r = idx % (CIN * Nd);
  int c = r / Nd, n = r % Nd;
  unsigned short h = f2bf(x[(size_t)idx * Td + t]);   // x layout (B,CIN,N,T)
  tem[((size_t)b * FEAT + c) * Nd + n] = h;
  gcatT[((size_t)b * Nd + n) * FK + c] = h;
}

// g_k = tem @ L_k^T for k=1,2; result stored transposed into gcatT[q][96k+f] (bf16)
__global__ void k_gemm1(const unsigned short* __restrict__ tem,
                        const unsigned short* __restrict__ L1,
                        const unsigned short* __restrict__ L2,
                        unsigned short* __restrict__ gcatT) {
  int lane = threadIdx.x & 31;
  int wave = threadIdx.x >> 5;
  int qblk = blockIdx.x;          // 0..7  (128 nodes each)
  int kL   = blockIdx.y;          // 0..1
  int b    = blockIdx.z;          // 0..15
  const unsigned short* Lm = (kL == 0) ? L1 : L2;
  const unsigned short* A  = tem + (size_t)b * FEAT * Nd;
  int q0 = qblk * 128 + wave * 16;
  v8f acc[6] = {};
  for (int k0 = 0; k0 < Nd; k0 += 32) {
    v16bf bt = load_tile_B(Lm, Nd, q0, k0, lane);   // B[n,q] = L[q][n]
    #pragma unroll
    for (int mt = 0; mt < 6; ++mt) {
      v16bf at = load_tile_A(A, Nd, mt * 16, k0, lane);
      acc[mt] = wmma_bf(at, bt, acc[mt]);
    }
  }
  int fkbase = 96 * (kL + 1);
  int q = q0 + (lane & 15);
  unsigned short* dst = gcatT + ((size_t)b * Nd + q) * FK + fkbase + ((lane < 16) ? 0 : 8);
  #pragma unroll
  for (int mt = 0; mt < 6; ++mt)
    *reinterpret_cast<uint4*>(dst + mt * 16) = pack8(acc[mt]);  // transposed store: per-lane contiguous
}

// fea = Wre @ gcat + b, fused LSTM pointwise; writes out, c, and h (bf16) for next step
__global__ void k_gemm2(const unsigned short* __restrict__ Wre,
                        unsigned short* gcatT,                   // read B, write h cols
                        const float* __restrict__ bias,
                        float* __restrict__ cstate,
                        unsigned short* __restrict__ tem,
                        float* __restrict__ out, int t) {
  __shared__ float fea[GATE][33];
  int lane = threadIdx.x & 31;
  int wave = threadIdx.x >> 5;
  int qblk = blockIdx.x;   // 0..31  (32 nodes each)
  int b    = blockIdx.y;
  int q0b  = qblk * 32;
  const unsigned short* Bbase = gcatT + (size_t)b * Nd * FK;
  v8f acc[2][2] = {};
  for (int k0 = 0; k0 < FK; k0 += 32) {
    v16bf bt0 = load_tile_B(Bbase, FK, q0b,      k0, lane);
    v16bf bt1 = load_tile_B(Bbase, FK, q0b + 16, k0, lane);
    #pragma unroll
    for (int i = 0; i < 2; ++i) {
      v16bf at = load_tile_A(Wre, FK, (wave * 2 + i) * 16, k0, lane);
      acc[i][0] = wmma_bf(at, bt0, acc[i][0]);
      acc[i][1] = wmma_bf(at, bt1, acc[i][1]);
    }
  }
  #pragma unroll
  for (int i = 0; i < 2; ++i)
    #pragma unroll
    for (int ct = 0; ct < 2; ++ct)
      #pragma unroll
      for (int r = 0; r < 8; ++r) {
        int row = (wave * 2 + i) * 16 + r + ((lane < 16) ? 0 : 8);
        int col = ct * 16 + (lane & 15);
        fea[row][col] = acc[i][ct][r] + bias[row];
      }
  __syncthreads();
  #pragma unroll
  for (int it = 0; it < 8; ++it) {
    int idx = threadIdx.x + it * 256;   // 0..2047  => 64 channels x 32 nodes
    int co = idx & 63;
    int qq = idx >> 6;
    int q  = q0b + qq;
    float iv = fea[co][qq];
    float jv = fea[co + 64][qq];
    float fv = fea[co + 128][qq];
    float ov = fea[co + 192][qq];
    size_t cidx = ((size_t)b * COUT + co) * Nd + q;
    float c  = cstate[cidx];
    float cn = c * sgm(fv) + sgm(iv) * tanhf(jv);
    float hn = tanhf(cn) * sgm(ov);
    cstate[cidx] = cn;
    out[cidx * Td + t] = hn;                                   // (B,COUT,N,T)
    unsigned short hb = f2bf(hn);
    tem[((size_t)b * FEAT + CIN + co) * Nd + q] = hb;          // h rows for next GEMM1
    gcatT[((size_t)b * Nd + q) * FK + CIN + co] = hb;          // h cols (k=0 block)
  }
}

extern "C" void kernel_launch(void* const* d_in, const int* in_sizes, int n_in,
                              void* d_out, int out_size, void* d_ws, size_t ws_size,
                              hipStream_t stream) {
  const float* x    = (const float*)d_in[0];
  const float* adj  = (const float*)d_in[1];
  const float* W    = (const float*)d_in[2];
  const float* bias = (const float*)d_in[3];
  float* out = (float*)d_out;

  char* ws = (char*)d_ws;
  size_t off = 0;
  auto alloc = [&](size_t bytes) -> char* {
    char* p = ws + off;
    off = (off + bytes + 255) & ~(size_t)255;
    return p;
  };
  unsigned short* adjbf = (unsigned short*)alloc((size_t)Nd * Nd * 2);      // = L1
  unsigned short* adjT  = (unsigned short*)alloc((size_t)Nd * Nd * 2);
  unsigned short* L2    = (unsigned short*)alloc((size_t)Nd * Nd * 2);
  unsigned short* Wre   = (unsigned short*)alloc((size_t)GATE * FK * 2);
  unsigned short* tem   = (unsigned short*)alloc((size_t)Bd * FEAT * Nd * 2);
  unsigned short* gcatT = (unsigned short*)alloc((size_t)Bd * Nd * FK * 2);
  float*          cst   = (float*)alloc((size_t)Bd * COUT * Nd * 4);

  k_setup_adj<<<(Nd * Nd + 255) / 256, 256, 0, stream>>>(adj, adjbf, adjT);
  k_setup_w<<<(GATE * FK + 255) / 256, 256, 0, stream>>>(W, Wre);
  k_cheb2<<<dim3(Nd / 128, Nd / 64), 256, 0, stream>>>(adjbf, adjT, L2);
  k_init<<<(Bd * COUT * Nd + 255) / 256, 256, 0, stream>>>(cst, tem, gcatT);

  for (int t = 0; t < Td; ++t) {
    k_prep<<<(Bd * CIN * Nd + 255) / 256, 256, 0, stream>>>(x, tem, gcatT, t);
    k_gemm1<<<dim3(8, 2, Bd), 256, 0, stream>>>(tem, adjbf, L2, gcatT);
    k_gemm2<<<dim3(32, Bd), 256, 0, stream>>>(Wre, gcatT, bias, cst, tem, out, t);
  }
}
